// Head_59502476919072
// MI455X (gfx1250) — matrix operand order repaired and checked by reading, not compiled, not run on hardware
//
#include <hip/hip_runtime.h>
#include <hip/hip_bf16.h>
#include <math.h>

// Problem constants (match reference)
#define BB 4
#define TT 4096
#define CC 1024
#define HH 128
#define SCALE 0.08838834764831845f  // 1/sqrt(128)

typedef __attribute__((ext_vector_type(16))) __bf16 v16bf;
typedef __attribute__((ext_vector_type(8)))  float  v8f;
typedef __attribute__((ext_vector_type(4))) unsigned int u32x4;
typedef __attribute__((ext_vector_type(8))) int i32x8;
typedef __attribute__((ext_vector_type(4))) int i32x4;

union AFrag {
  v16bf v;
  __bf16 e[16];
  uint4 q[2];  // two 16-byte halves (e[0..7], e[8..15])
};
union CFrag {
  v8f v;
  float e[8];
};
union Pack8 {
  uint4 u;
  __bf16 h[8];
};

static __device__ inline v8f wmma_bf16(v16bf a, v16bf b, v8f c) {
  // (neg_a, A, neg_b, B, c_mod, C, reuse_a, reuse_b)
  return __builtin_amdgcn_wmma_f32_16x16x32_bf16(false, a, false, b, (short)0, c,
                                                 false, false);
}

static __device__ inline v8f zero8() {
  v8f z = {0.f, 0.f, 0.f, 0.f, 0.f, 0.f, 0.f, 0.f};
  return z;
}

// ---------------------------------------------------------------------------
// Kernel 0: pack fp32 weights into bf16 *B-fragment layout*:
//   wp[((m*32 + kc)*8 + strip)*32 + lane][e]  =  W_m[kc*32 + 8*(lane>>4) + (e&7) + 16*(e>>3)]
//                                                   [strip*16 + (lane&15)]
// so each lane's 16 fragment elements are 32 contiguous bytes (2x b128 load).
// ---------------------------------------------------------------------------
__global__ __launch_bounds__(256) void pack_w(
    const float* __restrict__ Wq, const float* __restrict__ Wk,
    const float* __restrict__ Wv, __bf16* __restrict__ wp) {
  const int i = blockIdx.x * 256 + threadIdx.x;  // < 3*32*8*32*16 = 393216
  const int e  = i & 15;
  const int L  = (i >> 4) & 31;
  const int s  = (i >> 9) & 7;
  const int kc = (i >> 12) & 31;
  const int m  = i >> 17;  // 0..2
  const int kk  = kc * 32 + 8 * (L >> 4) + (e & 7) + 16 * (e >> 3);
  const int col = s * 16 + (L & 15);
  const float* W = (m == 0) ? Wq : (m == 1) ? Wk : Wv;
  wp[i] = (__bf16)W[(size_t)kk * HH + col];
}

// ---------------------------------------------------------------------------
// Kernel 1: QKV projection.  Block = 256 threads (8 waves).
// Block computes 32 rows x 128 cols of q,k,v.  Wave w owns N-strip w*16.
// ---------------------------------------------------------------------------
__global__ __launch_bounds__(256) void qkv_proj(
    const float* __restrict__ x, const __bf16* __restrict__ wp,
    const float* __restrict__ bq, const float* __restrict__ bk,
    const float* __restrict__ bv,
    __bf16* __restrict__ q, __bf16* __restrict__ k, __bf16* __restrict__ v) {
  __shared__ __align__(16) __bf16 xs[32][40];  // 32x32 K-chunk, padded rows

  const int tid  = threadIdx.x;
  const int wave = tid >> 5;
  const int lane = tid & 31;
  const int hi   = lane >> 4;
  const int ln   = lane & 15;
  const int m0   = blockIdx.x * 32;  // rows of flattened (B*T)
  const int n0   = wave * 16;        // N strip within H

  const int WMAT = 32 * 8 * 32 * 16;  // elements per packed matrix (131072)

  CFrag cq0, cq1, ck0, ck1, cv0, cv1;
  cq0.v = zero8(); cq1.v = zero8();
  ck0.v = zero8(); ck1.v = zero8();
  cv0.v = zero8(); cv1.v = zero8();

  for (int k0 = 0; k0 < CC; k0 += 32) {
    __syncthreads();  // protect xs from previous iteration's readers
    {
      // 256 threads load the 32x32 fp32 tile (one float4 each) -> bf16 LDS
      const int r  = tid >> 3;
      const int c4 = tid & 7;
      const float4 f = *reinterpret_cast<const float4*>(
          x + (size_t)(m0 + r) * CC + k0 + c4 * 4);
      xs[r][c4 * 4 + 0] = (__bf16)f.x;
      xs[r][c4 * 4 + 1] = (__bf16)f.y;
      xs[r][c4 * 4 + 2] = (__bf16)f.z;
      xs[r][c4 * 4 + 3] = (__bf16)f.w;
      if (k0 + 32 < CC)
        __builtin_prefetch(x + (size_t)(m0 + r) * CC + k0 + 32 + c4 * 4, 0, 3);
    }
    __syncthreads();

    // A fragments: each half is 8 consecutive bf16 in an LDS row (16B)
    AFrag a0, a1, fq, fk, fv;
    a0.q[0] = *reinterpret_cast<const uint4*>(&xs[ln][8 * hi]);
    a0.q[1] = *reinterpret_cast<const uint4*>(&xs[ln][16 + 8 * hi]);
    a1.q[0] = *reinterpret_cast<const uint4*>(&xs[16 + ln][8 * hi]);
    a1.q[1] = *reinterpret_cast<const uint4*>(&xs[16 + ln][16 + 8 * hi]);

    // Packed-W fragments: 32 contiguous bytes per lane (2x b128)
    const int kc = k0 >> 5;
    const size_t fo = ((size_t)(kc * 8 + wave) * 32 + lane) * 16;
    const uint4* wq4 = reinterpret_cast<const uint4*>(wp + fo);
    const uint4* wk4 = reinterpret_cast<const uint4*>(wp + WMAT + fo);
    const uint4* wv4 = reinterpret_cast<const uint4*>(wp + 2 * WMAT + fo);
    fq.q[0] = wq4[0]; fq.q[1] = wq4[1];
    fk.q[0] = wk4[0]; fk.q[1] = wk4[1];
    fv.q[0] = wv4[0]; fv.q[1] = wv4[1];

    cq0.v = wmma_bf16(a0.v, fq.v, cq0.v);
    cq1.v = wmma_bf16(a1.v, fq.v, cq1.v);
    ck0.v = wmma_bf16(a0.v, fk.v, ck0.v);
    ck1.v = wmma_bf16(a1.v, fk.v, ck1.v);
    cv0.v = wmma_bf16(a0.v, fv.v, cv0.v);
    cv1.v = wmma_bf16(a1.v, fv.v, cv1.v);
  }

  const float bql = bq[n0 + ln];
  const float bkl = bk[n0 + ln];
  const float bvl = bv[n0 + ln];
#pragma unroll
  for (int r = 0; r < 8; ++r) {
    const int row = r + 8 * hi;  // C/D layout
    const size_t o0 = (size_t)(m0 + row) * HH + n0 + ln;
    const size_t o1 = (size_t)(m0 + 16 + row) * HH + n0 + ln;
    q[o0] = (__bf16)(cq0.e[r] + bql);
    q[o1] = (__bf16)(cq1.e[r] + bql);
    k[o0] = (__bf16)(ck0.e[r] + bkl);
    k[o1] = (__bf16)(ck1.e[r] + bkl);
    v[o0] = (__bf16)(cv0.e[r] + bvl);
    v[o1] = (__bf16)(cv1.e[r] + bvl);
  }
}

// ---------------------------------------------------------------------------
// Kernel 2: causal flash attention.  Block = 128 threads (4 waves).
// Each wave owns 16 q rows (Q in A-fragments, O in fp32 accumulators).
// K tile is DMA'd by the Tensor Data Mover straight into padded LDS
// (tensor_load_to_lds + s_wait_tensorcnt); V is streamed through VGPRs and
// transposed into LDS so the P@V B-fragment is 2 contiguous b128 reads.
// ---------------------------------------------------------------------------
__global__ __launch_bounds__(128) void flash_attn(
    const __bf16* __restrict__ qb, const __bf16* __restrict__ kb,
    const __bf16* __restrict__ vb, float* __restrict__ out) {
  __shared__ __align__(16) __bf16 ks[32][136];   // K tile, row-major, padded
  __shared__ __align__(16) __bf16 vt[128][40];   // V tile transposed [h][s]
  __shared__ __align__(16) __bf16 ps[4][16][40]; // per-wave P staging (16x32)

  const int tid  = threadIdx.x;
  const int wave = tid >> 5;
  const int lane = tid & 31;
  const int hi   = lane >> 4;
  const int ln   = lane & 15;
  const int b    = blockIdx.y;
  const int qblk = blockIdx.x;              // 64-row q block
  const int q_row = qblk * 64 + wave * 16;  // wave's first q row (in batch)
  const size_t base = (size_t)b * TT * HH;

  // LDS byte offset of the K tile (for the TDM descriptor)
  const unsigned ks_lds = (unsigned)(uintptr_t)(&ks[0][0]);

  // Tensor descriptor group 1 (constant across iterations):
  //  data_size=2B (code 1); pad_enable; pad every 64 DWORDs (code 5) by
  //  4 DWORDs (code 3)  -> LDS row stride 272B == ks row stride.
  //  tensor_dim0 = 128 (bits 79:48), tensor_dim1 = 4096 (bits 111:80),
  //  tile_dim0 = 128 (bits 127:112), tile_dim1 = 32 (bits 143:128),
  //  tensor_dim0_stride = 128 (bits 207:160).
  const i32x8 g1 = {
      (int)((1u << 16) | (1u << 20) | (5u << 22) | (3u << 25)),  // d0
      (int)(128u << 16),   // d1: tensor_dim0 low 16
      (int)(4096u << 16),  // d2: tensor_dim0 hi=0, tensor_dim1 low 16
      (int)(128u << 16),   // d3: tensor_dim1 hi=0, tile_dim0=128
      (int)32,             // d4: tile_dim1=32, tile_dim2=0
      (int)128,            // d5: tensor_dim0_stride low 32
      0,                   // d6
      0};                  // d7
  const i32x4 gz4 = {0, 0, 0, 0};              // groups 2/3 unused (2-D tile)
  const i32x8 gz8 = {0, 0, 0, 0, 0, 0, 0, 0};  // trailing group unused

  // Load the wave's 16x128 Q tile as 4 A-fragments: 2 x b128 per K-chunk
  AFrag aq[4];
  {
    const __bf16* qrow = qb + base + (size_t)(q_row + ln) * HH;
#pragma unroll
    for (int c = 0; c < 4; ++c) {
      aq[c].q[0] = *reinterpret_cast<const uint4*>(qrow + 32 * c + 16 * hi);
      aq[c].q[1] = *reinterpret_cast<const uint4*>(qrow + 32 * c + 16 * hi + 8);
    }
  }

  CFrag o[8];
  float mrow[8], lrow[8];
#pragma unroll
  for (int i = 0; i < 8; ++i) {
    o[i].v  = zero8();
    mrow[i] = -INFINITY;
    lrow[i] = 0.f;
  }

  const int s_end  = qblk * 64 + 64;  // block's causal bound (exclusive)
  const int t_maxw = q_row + 15;      // wave's last q row

  for (int s0 = 0; s0 < s_end; s0 += 32) {
    __syncthreads();  // previous iteration's LDS readers are done
    if (wave == 0) {
      // TDM: DMA the 32x128 bf16 K tile into padded LDS rows
      const unsigned long long ga =
          (unsigned long long)(uintptr_t)(kb + base + (size_t)s0 * HH);
      u32x4 g0;
      g0.x = 1u;  // one valid descriptor
      g0.y = ks_lds;
      g0.z = (unsigned)ga;
      g0.w = (unsigned)((ga >> 32) & 0x1FFFFFFull) | (2u << 30);  // type=2
      __builtin_amdgcn_tensor_load_to_lds(g0, g1, gz4, gz4, gz8, 0);
    }
    {
      // V: 32x128 bf16 tile, transposed on the way into LDS
      const uint4* gv =
          reinterpret_cast<const uint4*>(vb + base + (size_t)s0 * HH);
#pragma unroll
      for (int i = 0; i < 4; ++i) {
        const int idx = tid + i * 128;  // 0..511 (16 uint4 per row)
        const int row = idx >> 4;       // s index 0..31
        const int c16 = idx & 15;       // h group (8 bf16 each)
        Pack8 pv;
        pv.u = gv[row * 16 + c16];
        const int h0 = c16 * 8;
#pragma unroll
        for (int j = 0; j < 8; ++j) vt[h0 + j][row] = pv.h[j];
      }
      if (s0 + 32 < s_end) {
        __builtin_prefetch(kb + base + (size_t)(s0 + 32) * HH + tid * 8, 0, 3);
        __builtin_prefetch(vb + base + (size_t)(s0 + 32) * HH + tid * 8, 0, 3);
      }
    }
    if (wave == 0) __builtin_amdgcn_s_wait_tensorcnt(0);
    __syncthreads();

    if (s0 <= t_maxw) {  // wave has unmasked work in this chunk
      // ---- S = Q @ K^T for two 16-col sub-blocks ----
      CFrag sS[2];
#pragma unroll
      for (int sbi = 0; sbi < 2; ++sbi) {
        sS[sbi].v = zero8();
#pragma unroll
        for (int c = 0; c < 4; ++c) {
          AFrag bk_;  // 16 consecutive h in one LDS row -> 2 x b128
          bk_.q[0] = *reinterpret_cast<const uint4*>(
              &ks[sbi * 16 + ln][32 * c + 16 * hi]);
          bk_.q[1] = *reinterpret_cast<const uint4*>(
              &ks[sbi * 16 + ln][32 * c + 16 * hi + 8]);
          sS[sbi].v = wmma_bf16(aq[c].v, bk_.v, sS[sbi].v);
        }
      }
      // ---- online softmax over the 32-wide chunk ----
      const bool need_mask = (s0 + 31 > q_row);  // chunk touches the diagonal
#pragma unroll
      for (int r = 0; r < 8; ++r) {
        float v0 = sS[0].e[r] * SCALE;
        float v1 = sS[1].e[r] * SCALE;
        if (need_mask) {
          const int t = q_row + r + 8 * hi;
          if (s0 + ln > t)      v0 = -INFINITY;
          if (s0 + 16 + ln > t) v1 = -INFINITY;
        }
        float rm = fmaxf(v0, v1);
#pragma unroll
        for (int mk = 1; mk < 16; mk <<= 1)
          rm = fmaxf(rm, __shfl_xor(rm, mk, 32));
        const float nm   = fmaxf(mrow[r], rm);
        const float corr = __expf(mrow[r] - nm);
        const float p0   = __expf(v0 - nm);
        const float p1   = __expf(v1 - nm);
        float psum = p0 + p1;
#pragma unroll
        for (int mk = 1; mk < 16; mk <<= 1)
          psum += __shfl_xor(psum, mk, 32);
        lrow[r] = lrow[r] * corr + psum;
        mrow[r] = nm;
#pragma unroll
        for (int st = 0; st < 8; ++st) o[st].e[r] *= corr;
        ps[wave][r + 8 * hi][ln]      = (__bf16)p0;
        ps[wave][r + 8 * hi][16 + ln] = (__bf16)p1;
      }
      // ---- O += P @ V ----
      AFrag ap;  // each half contiguous in the ps row -> 2 x b128
      ap.q[0] = *reinterpret_cast<const uint4*>(&ps[wave][ln][16 * hi]);
      ap.q[1] = *reinterpret_cast<const uint4*>(&ps[wave][ln][16 * hi + 8]);
#pragma unroll
      for (int st = 0; st < 8; ++st) {
        AFrag bv_;  // vt row = h (st*16+ln), cols s = 16*hi..16*hi+15
        bv_.q[0] = *reinterpret_cast<const uint4*>(&vt[st * 16 + ln][16 * hi]);
        bv_.q[1] =
            *reinterpret_cast<const uint4*>(&vt[st * 16 + ln][16 * hi + 8]);
        o[st].v = wmma_bf16(ap.v, bv_.v, o[st].v);
      }
    }
  }

  // Epilogue: out = O / l  (fp32 output)
#pragma unroll
  for (int st = 0; st < 8; ++st) {
#pragma unroll
    for (int r = 0; r < 8; ++r) {
      const int t = q_row + r + 8 * hi;
      out[base + (size_t)t * HH + st * 16 + ln] = o[st].e[r] / lrow[r];
    }
  }
}

// ---------------------------------------------------------------------------
// Launcher
// ---------------------------------------------------------------------------
extern "C" void kernel_launch(void* const* d_in, const int* in_sizes, int n_in,
                              void* d_out, int out_size, void* d_ws,
                              size_t ws_size, hipStream_t stream) {
  const float* x  = (const float*)d_in[0];
  const float* Wq = (const float*)d_in[1];
  const float* bq = (const float*)d_in[2];
  const float* Wk = (const float*)d_in[3];
  const float* bk = (const float*)d_in[4];
  const float* Wv = (const float*)d_in[5];
  const float* bv = (const float*)d_in[6];
  float* out = (float*)d_out;

  const size_t CH  = (size_t)CC * HH;       // 131072
  const size_t BTH = (size_t)BB * TT * HH;  // 2097152

  __bf16* w  = (__bf16*)d_ws;
  __bf16* wp = w;            // packed weights: 3*CH elements
  __bf16* qb = w + 3 * CH;
  __bf16* kb = qb + BTH;
  __bf16* vb = kb + BTH;

  pack_w<<<(int)((3 * CH) / 256), 256, 0, stream>>>(Wq, Wk, Wv, wp);
  qkv_proj<<<(BB * TT) / 32, 256, 0, stream>>>(x, wp, bq, bk, bv, qb, kb, vb);
  dim3 grid(TT / 64, BB);
  flash_attn<<<grid, 128, 0, stream>>>(qb, kb, vb, out);
}
